// LearnableSeriesDecomp_34359738368026
// MI455X (gfx1250) — compile-verified, matrix operand order checked by reading, not run on hardware
//
#include <hip/hip_runtime.h>
#include <hip/hip_bf16.h>

typedef __attribute__((ext_vector_type(16))) _Float16 v16h;
typedef __attribute__((ext_vector_type(2)))  _Float16 v2h;
typedef __attribute__((ext_vector_type(8)))  float    v8f;

#define B_   32
#define L_   2048
#define D_   512
#define H_   256
#define KMAX 50
#define PMAX 24
#define WEFF 49
#define LT   64
#define DT   64

// ---------------------------------------------------------------------------
// 1) xg[b,d] = mean_l x[b,l,d], stored directly as f16 (WMMA A operand)
// ---------------------------------------------------------------------------
__global__ void k_colmean(const float* __restrict__ x, _Float16* __restrict__ xgh) {
  int t = blockIdx.x * blockDim.x + threadIdx.x;      // 0 .. B*D-1
  int b = t / D_, d = t % D_;
  const float* p = x + (size_t)b * L_ * D_ + d;
  float s = 0.f;
  for (int l = 0; l < L_; ++l) s += p[(size_t)l * D_];
  xgh[t] = (_Float16)(s * (1.0f / L_));
}

// ---------------------------------------------------------------------------
// 2) W1 [H,D] fp32 -> f16 (WMMA B operand: B[k,n] = W1[n,k] — W1 row-major is
//    already the per-lane column layout indexed below)
// ---------------------------------------------------------------------------
__global__ void k_cvt_w1(const float* __restrict__ W1, _Float16* __restrict__ W1h, int n) {
  int t = blockIdx.x * blockDim.x + threadIdx.x;
  if (t < n) W1h[t] = (_Float16)W1[t];
}

// ---------------------------------------------------------------------------
// 3) h = relu(xg @ W1^T + b1)   M=32, N=256, K=512 via v_wmma_f32_16x16x32_f16
//    One block, 8 waves; 32 tiles of 16x16; each wave does 4 tiles.
// ---------------------------------------------------------------------------
__global__ void k_mlp_wmma(const _Float16* __restrict__ Ah,   // [32][512]
                           const _Float16* __restrict__ Bh,   // [256][512] (= W1 f16)
                           const float*    __restrict__ b1,
                           float*          __restrict__ hout) // [32][256]
{
  const int lane = threadIdx.x & 31;
  const int wv   = threadIdx.x >> 5;      // 0..7
  const int r    = lane & 15;
  const int hi   = lane >> 4;             // 0 or 1

  for (int it = 0; it < 4; ++it) {
    const int tI  = wv + 8 * it;          // 0..31
    const int m_t = tI >> 4;              // 0..1
    const int n_t = tI & 15;              // 0..15
    const int m   = m_t * 16 + r;
    const int n   = n_t * 16 + r;

    v8f acc = {};
    for (int kk = 0; kk < D_; kk += 32) {
      v16h a, b;
#pragma unroll
      for (int p = 0; p < 8; ++p) {
        // ISA 16-bit A-layout: VGPR p -> K = 2*(p&3) + (p>=4 ? 16:0) + (lane>=16 ? 8:0)
        const int k0 = kk + 2 * (p & 3) + ((p >> 2) << 4) + (hi << 3);
        v2h av = *reinterpret_cast<const v2h*>(Ah + m * D_ + k0);
        v2h bv = *reinterpret_cast<const v2h*>(Bh + n * D_ + k0);
        a[2 * p] = av[0]; a[2 * p + 1] = av[1];
        b[2 * p] = bv[0]; b[2 * p + 1] = bv[1];
      }
      acc = __builtin_amdgcn_wmma_f32_16x16x32_f16(
          /*neg_a=*/false, a, /*neg_b=*/false, b,
          /*c_mod=*/(short)0, acc, /*reuse_a=*/false, /*reuse_b=*/false);
    }

    // C layout: VGPR rr -> M = rr + 8*hi, N = lane&15
    const int   nc   = n_t * 16 + r;
    const float bias = b1[nc];
#pragma unroll
    for (int rr = 0; rr < 8; ++rr) {
      const int mm = m_t * 16 + rr + hi * 8;
      float v = acc[rr] + bias;
      hout[mm * H_ + nc] = v > 0.f ? v : 0.f;
    }
  }
}

// ---------------------------------------------------------------------------
// 4) logit = sigmoid(h @ W2^T + b2); k[b] = clamp/odd-force (round-half-even)
// ---------------------------------------------------------------------------
__global__ void k_head(const float* __restrict__ h, const float* __restrict__ W2,
                       const float* __restrict__ b2, int* __restrict__ kout) {
  int b = threadIdx.x;
  if (b >= B_) return;
  float s = b2[0];
  for (int j = 0; j < H_; ++j) s += h[b * H_ + j] * W2[j];
  float logit = 1.0f / (1.0f + __expf(-s));
  float kf = logit * (float)(KMAX - 5) + 5.0f;
  int k = (int)rintf(kf);                 // round-half-even like jnp.round
  k = min(k, KMAX);
  k = min(k, L_ / 2);
  k = max(k, 3);
  if ((k & 1) == 0) k -= 1;
  k = max(k, 3);
  kout[b] = k;
}

// ---------------------------------------------------------------------------
// 5) per-(b,d): softmax over first k taps, re-center into 49-wide window
// ---------------------------------------------------------------------------
__global__ void k_weff(const float* __restrict__ tw, const int* __restrict__ kb,
                       float* __restrict__ weff) {
  int t = blockIdx.x * blockDim.x + threadIdx.x;      // 0 .. B*D-1
  int b = t / D_, d = t % D_;
  int k = kb[b];                                      // uniform per wave
  const float* w = tw + d * KMAX;                     // trend_weights[d,0,:]
  float m = -1e30f;
  for (int j = 0; j < k; ++j) m = fmaxf(m, w[j]);
  float s = 0.f;
  for (int j = 0; j < k; ++j) s += __expf(w[j] - m);
  float inv = 1.0f / s;
  int shift = k >> 1;
  float* o = weff + (size_t)t * WEFF;
  for (int i = 0; i < WEFF; ++i) {
    int j = i - PMAX + shift;
    o[i] = (j >= 0 && j < k) ? __expf(w[j] - m) * inv : 0.f;
  }
}

// ---------------------------------------------------------------------------
// 6) depthwise conv with replicate pad; seasonal = x - trend
//    - x tile staged with gfx1250 async global->LDS (no VGPR round-trip)
//    - weights held in VGPRs; diagonal (s = i + w) reuse: one LDS x-read
//      feeds up to 16 FMAs  ->  113 LDS reads/thread instead of 833
// ---------------------------------------------------------------------------
__global__ void __launch_bounds__(256)
k_conv(const float* __restrict__ x, const float* __restrict__ weff,
       float* __restrict__ seasonal, float* __restrict__ trend) {
  __shared__ float xs[(LT + 2 * PMAX) * DT];          // 112*64 fp32 = 28 KB, bank = d
  __shared__ float wsh[DT * WEFF];                    // 64*49 fp32 = 12.25 KB

  const int b  = blockIdx.z;
  const int l0 = blockIdx.x * LT;
  const int d0 = blockIdx.y * DT;

  // ---- async stage x tile: 1792 x 16B transfers, 7 per thread ----
  for (int idx = threadIdx.x; idx < (LT + 2 * PMAX) * (DT / 4); idx += blockDim.x) {
    int row = idx >> 4;                               // /(DT/4)
    int c4  = (idx & 15) * 4;
    int l = l0 - PMAX + row;
    l = max(0, min(L_ - 1, l));                       // replicate pad
    const float* g = x + ((size_t)b * L_ + l) * D_ + d0 + c4;
    unsigned int lds_off = (unsigned int)(uintptr_t)(&xs[row * DT + c4]);
    asm volatile("global_load_async_to_lds_b128 %0, %1, off"
                 :: "v"(lds_off), "v"(g) : "memory");
  }
  // ---- weights: normal path ----
  for (int idx = threadIdx.x; idx < DT * WEFF; idx += blockDim.x) {
    int d = idx / WEFF, w = idx % WEFF;
    wsh[idx] = weff[((size_t)b * D_ + d0 + d) * WEFF + w];
  }
  asm volatile("s_wait_asynccnt 0x0" ::: "memory");
  __syncthreads();

  const int d  = threadIdx.x & (DT - 1);
  const int lb = (threadIdx.x >> 6) * 16;             // 4 l-groups of 16

  // all 49 taps in VGPRs
  float wt[WEFF];
#pragma unroll
  for (int w = 0; w < WEFF; ++w) wt[w] = wsh[d * WEFF + w];

  float acc[16];
#pragma unroll
  for (int i = 0; i < 16; ++i) acc[i] = 0.f;

  // diagonal register reuse: s = i + w, one LDS read per s
#pragma unroll
  for (int s = 0; s < 16 + WEFF - 1; ++s) {           // 0..63
    float xv = xs[(lb + s) * DT + d];
#pragma unroll
    for (int i = 0; i < 16; ++i) {
      int w = s - i;                                  // compile-time per (s,i)
      if (w >= 0 && w < WEFF)
        acc[i] = fmaf(xv, wt[w], acc[i]);
    }
  }

#pragma unroll
  for (int i = 0; i < 16; ++i) {
    int l = l0 + lb + i;
    float xv = xs[(lb + i + PMAX) * DT + d];
    size_t o = ((size_t)b * L_ + l) * D_ + d0 + d;
    trend[o]    = acc[i];
    seasonal[o] = xv - acc[i];
  }
}

// ---------------------------------------------------------------------------
extern "C" void kernel_launch(void* const* d_in, const int* in_sizes, int n_in,
                              void* d_out, int out_size, void* d_ws, size_t ws_size,
                              hipStream_t stream) {
  const float* x  = (const float*)d_in[0];   // [B,L,D]
  const float* tw = (const float*)d_in[1];   // [D,1,KMAX]
  const float* W1 = (const float*)d_in[2];   // [H,D]
  const float* b1 = (const float*)d_in[3];   // [H]
  const float* W2 = (const float*)d_in[4];   // [1,H]
  const float* b2 = (const float*)d_in[5];   // [1]

  float* seasonal = (float*)d_out;
  float* trend    = seasonal + (size_t)B_ * L_ * D_;

  // workspace layout (bytes)
  char* ws = (char*)d_ws;
  _Float16* xgh  = (_Float16*)(ws + 0);        //  32 KB  xg in f16 [32][512]
  _Float16* W1h  = (_Float16*)(ws + 32768);    // 256 KB  W1 in f16 [256][512]
  float*    hbuf = (float*)   (ws + 294912);   //  32 KB  h  [32][256]
  int*      kbuf = (int*)     (ws + 327680);   //  128 B  k  [32]
  float*    weff = (float*)   (ws + 327808);   // 3.2 MB  w_eff [32][512][49]

  k_colmean <<<B_ * D_ / 256, 256, 0, stream>>>(x, xgh);
  k_cvt_w1  <<<(H_ * D_ + 255) / 256, 256, 0, stream>>>(W1, W1h, H_ * D_);
  k_mlp_wmma<<<1, 256, 0, stream>>>(xgh, W1h, b1, hbuf);
  k_head    <<<1, 32, 0, stream>>>(hbuf, W2, b2, kbuf);
  k_weff    <<<B_ * D_ / 256, 256, 0, stream>>>(tw, kbuf, weff);
  k_conv    <<<dim3(L_ / LT, D_ / DT, B_), 256, 0, stream>>>(x, weff, seasonal, trend);
}
